// Scheduler_85993835200955
// MI455X (gfx1250) — matrix operand order repaired and checked by reading, not compiled, not run on hardware
//
#include <hip/hip_runtime.h>
#include <hip/hip_bf16.h>
#include <math.h>

#define BB   64
#define NOP  5000
#define NMA  128
#define NJO  100
#define FOP  8
#define DD   64
#define LATN 64
#define EPSV 1e-5f

typedef __attribute__((ext_vector_type(16))) _Float16 v16h;
typedef __attribute__((ext_vector_type(8)))  _Float16 v8h;
typedef __attribute__((ext_vector_type(8)))  float    v8f;

// ---------------------------------------------------------------- utilities
__global__ void k_zero(float* p, int n) {
    int i = blockIdx.x * blockDim.x + threadIdx.x;
    if (i < n) p[i] = 0.0f;
}

// ------------------------------------------------- ragged ope stats (per b)
__global__ void k_ope_stats(const float* __restrict__ raw_opes,
                            const int* __restrict__ nums,
                            float* __restrict__ mean_o, float* __restrict__ deno_o) {
    int b = blockIdx.x, tid = threadIdx.x;
    int n = nums[b];
    float s[8], sq[8];
    #pragma unroll
    for (int f = 0; f < 8; f++) { s[f] = 0.f; sq[f] = 0.f; }
    const float* base = raw_opes + (size_t)b * NOP * FOP;
    for (int r = tid; r < n; r += blockDim.x) {
        const float4* p = reinterpret_cast<const float4*>(base + (size_t)r * FOP);
        float4 a = p[0], c = p[1];
        float v[8] = {a.x, a.y, a.z, a.w, c.x, c.y, c.z, c.w};
        #pragma unroll
        for (int f = 0; f < 8; f++) { s[f] += v[f]; sq[f] += v[f] * v[f]; }
    }
    __shared__ float red[256 * 16];
    #pragma unroll
    for (int f = 0; f < 8; f++) { red[tid * 16 + f] = s[f]; red[tid * 16 + 8 + f] = sq[f]; }
    __syncthreads();
    for (int off = 128; off > 0; off >>= 1) {
        if (tid < off)
            for (int f = 0; f < 16; f++) red[tid * 16 + f] += red[(tid + off) * 16 + f];
        __syncthreads();
    }
    if (tid < 8) {
        float cnt  = (float)n;
        float mean = red[tid] / cnt;
        float var  = (red[8 + tid] - cnt * mean * mean) / (cnt - 1.0f);
        mean_o[b * 8 + tid] = mean;
        deno_o[b * 8 + tid] = sqrtf(var) + EPSV;
    }
}

// ------------------------------------------ nonzero proc_time stats (per b)
__global__ void k_proc_stats(const float* __restrict__ proc,
                             float* __restrict__ mean_p, float* __restrict__ deno_p) {
    int b = blockIdx.x, tid = threadIdx.x;
    const float4* p = reinterpret_cast<const float4*>(proc + (size_t)b * NOP * NMA);
    const int n4 = NOP * NMA / 4;
    float cnt = 0.f, s = 0.f, sq = 0.f;
    for (int i = tid; i < n4; i += blockDim.x) {
        float4 v = p[i];
        float vals[4] = {v.x, v.y, v.z, v.w};
        #pragma unroll
        for (int k = 0; k < 4; k++) {
            float x = vals[k];
            if (x != 0.0f) { cnt += 1.0f; s += x; sq += x * x; }
        }
    }
    __shared__ float rc[256], rs[256], rq[256];
    rc[tid] = cnt; rs[tid] = s; rq[tid] = sq;
    __syncthreads();
    for (int off = 128; off > 0; off >>= 1) {
        if (tid < off) { rc[tid] += rc[tid + off]; rs[tid] += rs[tid + off]; rq[tid] += rq[tid + off]; }
        __syncthreads();
    }
    if (tid == 0) {
        float mean = rs[0] / rc[0];
        float var  = (rq[0] - 2.0f * mean * rs[0] + rc[0] * mean * mean) / (rc[0] - 1.0f);
        mean_p[b] = mean;
        deno_p[b] = sqrtf(var) + EPSV;
    }
}

// -------------------------- machine norm + h_ma = norm_mas @ W_ma  (per b)
__global__ void k_hma(const float* __restrict__ raw_mas, const float* __restrict__ W_ma,
                      float* __restrict__ hma32, _Float16* __restrict__ hma16) {
    int b = blockIdx.x, m = threadIdx.x;           // 128 threads, one per machine
    __shared__ float sw[8 * 64];
    __shared__ float red[128 * 16];
    __shared__ float smean[8], sden[8];
    for (int i = m; i < 512; i += 128) sw[i] = W_ma[i];
    const float* row = raw_mas + ((size_t)b * NMA + m) * FOP;
    float v[8];
    #pragma unroll
    for (int f = 0; f < 8; f++) v[f] = row[f];
    #pragma unroll
    for (int f = 0; f < 8; f++) { red[m * 16 + f] = v[f]; red[m * 16 + 8 + f] = v[f] * v[f]; }
    __syncthreads();
    for (int off = 64; off > 0; off >>= 1) {
        if (m < off)
            for (int f = 0; f < 16; f++) red[m * 16 + f] += red[(m + off) * 16 + f];
        __syncthreads();
    }
    if (m < 8) {
        float cnt  = (float)NMA;
        float mean = red[m] / cnt;
        float var  = (red[8 + m] - cnt * mean * mean) / (cnt - 1.0f);
        smean[m] = mean;
        sden[m]  = sqrtf(var) + EPSV;
    }
    __syncthreads();
    float nm[8];
    #pragma unroll
    for (int f = 0; f < 8; f++) nm[f] = (v[f] - smean[f]) / sden[f];
    float*    o32 = hma32 + ((size_t)b * NMA + m) * DD;
    _Float16* o16 = hma16 + ((size_t)b * NMA + m) * DD;
    for (int d = 0; d < DD; d++) {
        float h = 0.f;
        #pragma unroll
        for (int f = 0; f < 8; f++) h += nm[f] * sw[f * 64 + d];
        o32[d] = h;
        o16[d] = (_Float16)h;
    }
}

// ------- main fused kernel: msg GEMM via WMMA + tanh epilogue + pooled sum
// A tile (adj .* proc_norm, f16) staged once per block in LDS; 4 waves cover
// the 4 column tiles of D=64, so proc/adj global traffic is exactly 1x.
__global__ void __launch_bounds__(128)
k_msg_wmma(const float* __restrict__ proc, const int* __restrict__ adj,
           const float* __restrict__ raw_opes,
           const float* __restrict__ W_ope, const float* __restrict__ b_ope,
           const int* __restrict__ nums,
           const float* __restrict__ mean_o, const float* __restrict__ deno_o,
           const float* __restrict__ mean_p, const float* __restrict__ deno_p,
           const _Float16* __restrict__ hma16,
           float* __restrict__ pooled_sum) {
    const int  b      = blockIdx.y;
    const int  tid    = threadIdx.x;
    const int  wave   = tid >> 5;        // 4 waves -> 4 n-tiles of 16 columns
    const int  lane   = tid & 31;
    const int  laneLo = lane & 15;
    const bool hi     = lane >= 16;
    const int  n0     = wave * 16;

    __shared__ float    sW[8 * 64];
    __shared__ float    sB[64];
    __shared__ float    sNo[16 * 8];
    __shared__ float    sMo[8], sDo[8];
    __shared__ float    sPool[128];
    __shared__ _Float16 sAh[16 * 128];   // staged normalized A tile (f16)

    for (int i = tid; i < 512; i += 128) sW[i] = W_ope[i];
    if (tid < 64) sB[tid] = b_ope[tid];
    if (tid < 8) { sMo[tid] = mean_o[b * 8 + tid]; sDo[tid] = 1.0f / deno_o[b * 8 + tid]; }
    __syncthreads();

    const float mp  = mean_p[b];
    const float rdp = 1.0f / deno_p[b];
    const int   nb  = nums[b];

    // --- preload B fragments: h_ma slice [128 x 16] as 4 k-step v16h regs
    v16h bfrag[4];
    {
        const _Float16* hb = hma16 + (size_t)b * NMA * DD;
        const int kb  = hi ? 8 : 0;
        const int col = n0 + laneLo;
        #pragma unroll
        for (int s = 0; s < 4; s++)
            #pragma unroll
            for (int e = 0; e < 16; e++)
                bfrag[s][e] = hb[(size_t)(s * 32 + kb + (e & 7) + (e >> 3) * 16) * DD + col];
    }

    float acc = 0.0f;
    const int nTiles = (NOP + 15) / 16;           // 313
    const int tilesPerBlock = 8;
    const int t0 = blockIdx.x * tilesPerBlock;

    for (int ti = 0; ti < tilesPerBlock; ti++) {
        int tile = t0 + ti;
        if (tile >= nTiles) break;                // block-uniform
        int rowBase = tile * 16;

        __syncthreads();                          // protect sNo/sAh from prev iter readers
        // stage normalized ope features for the 16 rows of this tile
        {
            int r = tid >> 3, f = tid & 7;
            int grow = rowBase + r;
            float val = 0.0f;
            if (grow < NOP)
                val = (raw_opes[((size_t)b * NOP + grow) * FOP + f] - sMo[f]) * sDo[f];
            sNo[r * 8 + f] = val;
        }
        // stage A tile: thread t covers row t>>3, cols (t&7)*16 .. +15
        {
            int r  = tid >> 3;
            int c0 = (tid & 7) * 16;
            int grow = rowBase + r;
            int crow = grow < NOP ? grow : (NOP - 1);      // branch-free clamp
            float rowValid = (grow < NOP) ? 1.0f : 0.0f;
            const float* prow  = proc + ((size_t)b * NOP + crow) * NMA + c0;
            const int*   adjrw = adj  + ((size_t)b * NOP + crow) * NMA + c0;
            #pragma unroll
            for (int q = 0; q < 4; q++) {
                float4 pv = reinterpret_cast<const float4*>(prow)[q];
                int4   av = reinterpret_cast<const int4*>(adjrw)[q];
                float xs[4] = {pv.x, pv.y, pv.z, pv.w};
                int   as[4] = {av.x, av.y, av.z, av.w};
                #pragma unroll
                for (int k = 0; k < 4; k++) {
                    float x  = xs[k];
                    float an = (as[k] != 0 && x != 0.0f) ? (x - mp) * rdp : 0.0f;
                    sAh[r * 128 + c0 + q * 4 + k] = (_Float16)(an * rowValid);
                }
            }
        }
        __syncthreads();

        // WMMA: A fragments from LDS (two 16B ds loads per k-step)
        v8f c = {0, 0, 0, 0, 0, 0, 0, 0};
        #pragma unroll
        for (int s = 0; s < 4; s++) {
            int kb = s * 32 + (hi ? 8 : 0);       // A-layout: lanes 0-15 K {0..7,16..23}
            v8h alo = *reinterpret_cast<const v8h*>(&sAh[laneLo * 128 + kb]);
            v8h ahi = *reinterpret_cast<const v8h*>(&sAh[laneLo * 128 + kb + 16]);
            v16h afrag;
            #pragma unroll
            for (int e = 0; e < 8; e++) { afrag[e] = alo[e]; afrag[8 + e] = ahi[e]; }
            c = __builtin_amdgcn_wmma_f32_16x16x32_f16(false, afrag, false, bfrag[s],
                                                       (short)0, c, false, false);
        }

        // epilogue: h = tanh(norm_opes @ W_ope + b_ope + msg); ragged pooled sum
        const int col = n0 + laneLo;
        #pragma unroll
        for (int r = 0; r < 8; r++) {
            int ml   = r + (hi ? 8 : 0);          // C layout: lanes16-31 hold M+8
            int grow = rowBase + ml;
            float dot = sB[col];
            #pragma unroll
            for (int f = 0; f < 8; f++) dot += sNo[ml * 8 + f] * sW[f * 64 + col];
            float h = tanhf(dot + c[r]);
            acc += (grow < nb) ? h : 0.0f;
        }
    }

    __syncthreads();
    sPool[tid] = acc;
    __syncthreads();
    if (!hi) {
        float v = sPool[tid] + sPool[tid + 16];
        atomicAdd(&pooled_sum[b * DD + n0 + laneLo], v);
    }
}

// --------------------------------------------------------- ope_step_c rows
__global__ void k_rowc(const int* __restrict__ ope_step, const int* __restrict__ end_b,
                       int* __restrict__ rowc, float* __restrict__ out_rowc) {
    int i = blockIdx.x * blockDim.x + threadIdx.x;
    if (i < BB * NJO) {
        int s = ope_step[i], e = end_b[i];
        int r = s > e ? e : s;
        rowc[i] = r;
        out_rowc[i] = (float)r;
    }
}

// ----------------- recompute h_opes only for the gathered job rows (b,j,d)
__global__ void k_hjobs(const float* __restrict__ proc, const int* __restrict__ adj,
                        const float* __restrict__ raw_opes,
                        const float* __restrict__ W_ope, const float* __restrict__ b_ope,
                        const float* __restrict__ mean_o, const float* __restrict__ deno_o,
                        const float* __restrict__ mean_p, const float* __restrict__ deno_p,
                        const int* __restrict__ rowc,
                        const float* __restrict__ hma32,
                        float* __restrict__ hjobs) {
    int bj = blockIdx.x;
    int b  = bj / NJO;
    int d  = threadIdx.x;                 // 64 threads
    int row = rowc[bj];
    __shared__ float sa[128];
    __shared__ float sno[8];
    float mp = mean_p[b], rdp = 1.0f / deno_p[b];
    const float* prow  = proc + ((size_t)b * NOP + row) * NMA;
    const int*   adjrw = adj  + ((size_t)b * NOP + row) * NMA;
    for (int m = d; m < 128; m += 64) {
        float x = prow[m];
        sa[m] = (adjrw[m] != 0 && x != 0.0f) ? (x - mp) * rdp : 0.0f;
    }
    if (d < 8)
        sno[d] = (raw_opes[((size_t)b * NOP + row) * FOP + d] - mean_o[b * 8 + d])
                 / deno_o[b * 8 + d];
    __syncthreads();
    float msg = 0.f;
    const float* hb = hma32 + (size_t)b * NMA * DD + d;
    for (int m = 0; m < 128; m++) msg += sa[m] * hb[(size_t)m * DD];
    float dot = b_ope[d];
    #pragma unroll
    for (int f = 0; f < 8; f++) dot += sno[f] * W_ope[f * 64 + d];
    hjobs[(size_t)bj * DD + d] = tanhf(dot + msg);
}

// ------------------------------------------------ finalize pooled embedding
__global__ void k_pool(const float* __restrict__ pooled_sum, const int* __restrict__ nums,
                       float* __restrict__ pooled, float* __restrict__ out_pool) {
    int b = blockIdx.x, d = threadIdx.x;
    float v = pooled_sum[b * DD + d] / (float)nums[b];
    pooled[b * DD + d]   = v;
    out_pool[b * DD + d] = v;
}

// ------------------- actor MLP: score per (b,j) (score is machine-invariant)
__global__ void k_scores(const float* __restrict__ hjobs, const float* __restrict__ pooled,
                         const float* __restrict__ W1, const float* __restrict__ b1,
                         const float* __restrict__ W2, const float* __restrict__ b2,
                         const float* __restrict__ W3, const float* __restrict__ b3,
                         float* __restrict__ scorej) {
    int bj = blockIdx.x, b = bj / NJO, t = threadIdx.x;   // 64 threads
    __shared__ float x[128];
    __shared__ float h1[64];
    __shared__ float red[64];
    x[t]      = hjobs[(size_t)bj * DD + t];
    x[64 + t] = pooled[b * DD + t];
    __syncthreads();
    float a = b1[t];
    for (int k = 0; k < 128; k++) a += x[k] * W1[k * LATN + t];
    h1[t] = tanhf(a);
    __syncthreads();
    float a2 = b2[t];
    for (int k = 0; k < 64; k++) a2 += h1[k] * W2[k * LATN + t];
    float h2 = tanhf(a2);
    red[t] = h2 * W3[t];
    __syncthreads();
    for (int off = 32; off > 0; off >>= 1) {
        if (t < off) red[t] += red[t + off];
        __syncthreads();
    }
    if (t == 0) scorej[bj] = red[0] + b3[0];
}

// ------------------------------------ masked softmax over (m,j) per batch
__global__ void k_softmax(const float* __restrict__ scorej, const int* __restrict__ rowc,
                          const int* __restrict__ adj,
                          const unsigned char* __restrict__ m_ma,
                          const unsigned char* __restrict__ m_jp,
                          const unsigned char* __restrict__ m_jf,
                          float* __restrict__ out) {
    int b = blockIdx.x, tid = threadIdx.x;                // 256 threads
    __shared__ float sS[NJO];
    __shared__ int   sR[NJO];
    __shared__ unsigned char sJok[NJO];
    __shared__ unsigned char sM[NMA];
    __shared__ float red[256];
    for (int j = tid; j < NJO; j += 256) {
        sS[j]   = scorej[b * NJO + j];
        sR[j]   = rowc[b * NJO + j];
        sJok[j] = (m_jp[b * NJO + j] | m_jf[b * NJO + j]) ? 0 : 1;
    }
    for (int m = tid; m < NMA; m += 256) sM[m] = m_ma[b * NMA + m] ? 0 : 1;
    __syncthreads();
    const int tot = NMA * NJO;
    const int* adjb = adj + (size_t)b * NOP * NMA;
    float mx = -INFINITY;
    for (int i = tid; i < tot; i += 256) {
        int m = i / NJO, j = i - m * NJO;
        bool e = sJok[j] && sM[m] && (adjb[(size_t)sR[j] * NMA + m] == 1);
        if (e && sS[j] > mx) mx = sS[j];
    }
    red[tid] = mx;
    __syncthreads();
    for (int off = 128; off > 0; off >>= 1) {
        if (tid < off) red[tid] = fmaxf(red[tid], red[tid + off]);
        __syncthreads();
    }
    float gmax = red[0];
    __syncthreads();
    float s = 0.f;
    for (int i = tid; i < tot; i += 256) {
        int m = i / NJO, j = i - m * NJO;
        bool e = sJok[j] && sM[m] && (adjb[(size_t)sR[j] * NMA + m] == 1);
        if (e) s += expf(sS[j] - gmax);
    }
    red[tid] = s;
    __syncthreads();
    for (int off = 128; off > 0; off >>= 1) {
        if (tid < off) red[tid] += red[tid + off];
        __syncthreads();
    }
    float gsum = red[0];
    float inv  = gsum > 0.f ? 1.0f / gsum : 0.0f;
    for (int i = tid; i < tot; i += 256) {
        int m = i / NJO, j = i - m * NJO;
        bool e = sJok[j] && sM[m] && (adjb[(size_t)sR[j] * NMA + m] == 1);
        out[(size_t)b * tot + i] = e ? expf(sS[j] - gmax) * inv : 0.0f;
    }
}

// --------------------------------------------------------------- launcher
extern "C" void kernel_launch(void* const* d_in, const int* in_sizes, int n_in,
                              void* d_out, int out_size, void* d_ws, size_t ws_size,
                              hipStream_t stream) {
    const float* raw_opes = (const float*)d_in[0];
    const float* raw_mas  = (const float*)d_in[1];
    const float* proc     = (const float*)d_in[2];
    const float* W_ope    = (const float*)d_in[3];
    const float* b_ope    = (const float*)d_in[4];
    const float* W_ma     = (const float*)d_in[5];
    const float* W1       = (const float*)d_in[6];
    const float* b1       = (const float*)d_in[7];
    const float* W2       = (const float*)d_in[8];
    const float* b2       = (const float*)d_in[9];
    const float* W3       = (const float*)d_in[10];
    const float* b3       = (const float*)d_in[11];
    const int*   nums     = (const int*)d_in[12];
    const int*   ope_step = (const int*)d_in[13];
    const int*   end_b    = (const int*)d_in[14];
    const int*   adj      = (const int*)d_in[15];
    const unsigned char* m_ma = (const unsigned char*)d_in[16];
    const unsigned char* m_jp = (const unsigned char*)d_in[17];
    const unsigned char* m_jf = (const unsigned char*)d_in[18];

    char* wsb = (char*)d_ws;
    size_t off = 0;
    auto alloc = [&](size_t bytes) -> void* {
        void* p = wsb + off;
        off += (bytes + 255) & ~(size_t)255;
        return p;
    };
    float*    mean_o     = (float*)alloc(BB * 8 * sizeof(float));
    float*    deno_o     = (float*)alloc(BB * 8 * sizeof(float));
    float*    mean_p     = (float*)alloc(BB * sizeof(float));
    float*    deno_p     = (float*)alloc(BB * sizeof(float));
    float*    pooled_sum = (float*)alloc(BB * DD * sizeof(float));
    float*    pooled     = (float*)alloc(BB * DD * sizeof(float));
    float*    hma32      = (float*)alloc((size_t)BB * NMA * DD * sizeof(float));
    _Float16* hma16      = (_Float16*)alloc((size_t)BB * NMA * DD * sizeof(_Float16));
    float*    hjobs      = (float*)alloc((size_t)BB * NJO * DD * sizeof(float));
    float*    scorej     = (float*)alloc(BB * NJO * sizeof(float));
    int*      rowc       = (int*)alloc(BB * NJO * sizeof(int));

    float* out       = (float*)d_out;
    float* out_probs = out;                                  // [B, NM*NJ]
    float* out_rowc  = out + (size_t)BB * NMA * NJO;         // [B, NJ] as float
    float* out_pool  = out_rowc + BB * NJO;                  // [B, D]

    k_zero<<<(BB * DD + 255) / 256, 256, 0, stream>>>(pooled_sum, BB * DD);
    k_ope_stats<<<BB, 256, 0, stream>>>(raw_opes, nums, mean_o, deno_o);
    k_proc_stats<<<BB, 256, 0, stream>>>(proc, mean_p, deno_p);
    k_hma<<<BB, 128, 0, stream>>>(raw_mas, W_ma, hma32, hma16);

    dim3 g4(((NOP + 15) / 16 + 7) / 8, BB);                  // 40 x 64 blocks
    k_msg_wmma<<<g4, 128, 0, stream>>>(proc, adj, raw_opes, W_ope, b_ope, nums,
                                       mean_o, deno_o, mean_p, deno_p,
                                       hma16, pooled_sum);

    k_rowc<<<(BB * NJO + 255) / 256, 256, 0, stream>>>(ope_step, end_b, rowc, out_rowc);
    k_hjobs<<<BB * NJO, 64, 0, stream>>>(proc, adj, raw_opes, W_ope, b_ope,
                                         mean_o, deno_o, mean_p, deno_p,
                                         rowc, hma32, hjobs);
    k_pool<<<BB, 64, 0, stream>>>(pooled_sum, nums, pooled, out_pool);
    k_scores<<<BB * NJO, 64, 0, stream>>>(hjobs, pooled, W1, b1, W2, b2, W3, b3, scorej);
    k_softmax<<<BB, 256, 0, stream>>>(scorej, rowc, adj, m_ma, m_jp, m_jf, out_probs);
}
